// SparseMoERouter_62818191671490
// MI455X (gfx1250) — matrix operand order repaired and checked by reading, not compiled, not run on hardware
//
#include <hip/hip_runtime.h>
#include <cstdint>
#include <cstddef>

#define NTOK 16384
#define DDIM 1024
#define HDIM 4096
#define NEXP 8

typedef __bf16 v16bf __attribute__((ext_vector_type(16)));
typedef __bf16 v8bf  __attribute__((ext_vector_type(8)));
typedef __bf16 v4bf  __attribute__((ext_vector_type(4)));
typedef float  v8f   __attribute__((ext_vector_type(8)));

// ---------------- WMMA helper (bf16 -> f32 accumulate) ----------------
__device__ __forceinline__ v8f wmma_bf16(v8bf alo, v8bf ahi, v8bf blo, v8bf bhi, v8f c) {
  v16bf a = __builtin_shufflevector(alo, ahi, 0,1,2,3,4,5,6,7,8,9,10,11,12,13,14,15);
  v16bf b = __builtin_shufflevector(blo, bhi, 0,1,2,3,4,5,6,7,8,9,10,11,12,13,14,15);
  return __builtin_amdgcn_wmma_f32_16x16x32_bf16(false, a, false, b, (short)0, c, false, false);
}

// ---------------- async 16B global -> LDS copy (CDNA5 async path) ----------------
__device__ __forceinline__ void async_b128(void* lds, const void* g) {
  uint32_t lo = (uint32_t)(uintptr_t)lds;   // low 32 bits of generic LDS addr == LDS offset
  asm volatile("global_load_async_to_lds_b128 %0, %1, off" :: "v"(lo), "v"(g) : "memory");
}
__device__ __forceinline__ void wait_async0() {
  asm volatile("s_wait_asynccnt 0" ::: "memory");
}

// ---------------- utility kernels ----------------
__global__ void k_zero_f4(float4* __restrict__ p) {
  p[(size_t)blockIdx.x * 256 + threadIdx.x] = float4{0.f, 0.f, 0.f, 0.f};
}

__global__ void k_zero_counts(int* __restrict__ counts) {
  if (threadIdx.x < NEXP) counts[threadIdx.x] = 0;
}

// f32 -> bf16 straight convert (4 elements / thread)
__global__ void k_cvt(const float* __restrict__ src, uint16_t* __restrict__ dst) {
  size_t i = (size_t)blockIdx.x * 256 + threadIdx.x;
  float4 v = ((const float4*)src)[i];
  v4bf o = { (__bf16)v.x, (__bf16)v.y, (__bf16)v.z, (__bf16)v.w };
  ((v4bf*)dst)[i] = o;
}

// f32 [R][C] -> bf16 [C][R] tiled transpose-convert, per expert (blockIdx.z)
__global__ void k_tcvt(const float* __restrict__ src, uint16_t* __restrict__ dst, int R, int C) {
  __shared__ float tile[32][33];
  const float* s = src + (size_t)blockIdx.z * R * C;
  uint16_t*    d = dst + (size_t)blockIdx.z * R * C;
  int c0 = blockIdx.x * 32, r0 = blockIdx.y * 32;
  for (int i = threadIdx.y; i < 32; i += 8)
    tile[i][threadIdx.x] = s[(size_t)(r0 + i) * C + c0 + threadIdx.x];
  __syncthreads();
  for (int i = threadIdx.y; i < 32; i += 8) {
    union { __bf16 b; uint16_t u; } cv;
    cv.b = (__bf16)tile[threadIdx.x][i];
    d[(size_t)(c0 + i) * R + r0 + threadIdx.x] = cv.u;
  }
}

// ---------------- router: logits, softmax, top-2, renorm, scatter ----------------
__global__ void k_router(const float* __restrict__ x, const float* __restrict__ gw,
                         float* __restrict__ combine, int* __restrict__ counts,
                         int* __restrict__ lists) {
  const int tid = threadIdx.x;
  const int wave = tid >> 5, lane = tid & 31;
  const int t = blockIdx.x * 8 + wave;
  const float* xr = x + (size_t)t * DDIM;
  float acc[8];
#pragma unroll
  for (int e = 0; e < 8; ++e) acc[e] = 0.f;
  for (int d = lane; d < DDIM; d += 32) {
    float xv = xr[d];
    const float4* g = (const float4*)(gw + (size_t)d * 8);
    float4 g0 = g[0], g1 = g[1];
    acc[0] += xv * g0.x; acc[1] += xv * g0.y; acc[2] += xv * g0.z; acc[3] += xv * g0.w;
    acc[4] += xv * g1.x; acc[5] += xv * g1.y; acc[6] += xv * g1.z; acc[7] += xv * g1.w;
  }
#pragma unroll
  for (int e = 0; e < 8; ++e)
#pragma unroll
    for (int off = 16; off > 0; off >>= 1)
      acc[e] += __shfl_xor(acc[e], off, 32);
  if (lane == 0) {
    float mx = acc[0];
    for (int e = 1; e < 8; ++e) mx = fmaxf(mx, acc[e]);
    float p[8], s = 0.f;
    for (int e = 0; e < 8; ++e) { p[e] = __expf(acc[e] - mx); s += p[e]; }
    int e1 = 0;
    for (int e = 1; e < 8; ++e) if (p[e] > p[e1]) e1 = e;
    int e2 = (e1 == 0) ? 1 : 0;
    for (int e = 0; e < 8; ++e) if (e != e1 && p[e] > p[e2]) e2 = e;
    float v1 = p[e1] / s, v2 = p[e2] / s;
    float rs = 1.0f / (v1 + v2);
    float r1 = v1 * rs, r2 = v2 * rs;
    for (int e = 0; e < 8; ++e)
      combine[(size_t)t * 8 + e] = (e == e1) ? r1 : ((e == e2) ? r2 : 0.f);
    int p1 = atomicAdd(&counts[e1], 1); lists[e1 * NTOK + p1] = t;
    int p2 = atomicAdd(&counts[e2], 1); lists[e2 * NTOK + p2] = t;
  }
}

// ---------------- fused expert FFN: 32-token tiles, WMMA bf16 ----------------
__global__ __launch_bounds__(256, 1) void k_moe(
    const uint16_t* __restrict__ xb_, const uint16_t* __restrict__ w1b_,
    const uint16_t* __restrict__ w2b_, const float* __restrict__ combine,
    const int* __restrict__ counts, const int* __restrict__ lists,
    float* __restrict__ out) {
  __shared__ __attribute__((aligned(16))) __bf16 Xs[32][1032];  // tokens x K(D), padded
  __shared__ __attribute__((aligned(16))) __bf16 Hs[32][40];    // tokens x Hc, padded
  __shared__ float Red[6][512];                                 // GEMM1 K-split partials
  __shared__ int   Toks[32];
  __shared__ float Scl[32];

  const __bf16* xb  = (const __bf16*)xb_;
  const __bf16* w1b = (const __bf16*)w1b_;   // [E][H][D] (transposed, bf16)
  const __bf16* w2b = (const __bf16*)w2b_;   // [E][D][H] (transposed, bf16)

  const int e    = blockIdx.x >> 9;
  const int tile = blockIdx.x & 511;
  const int cnt  = counts[e];
  const int row0 = tile << 5;
  if (row0 >= cnt) return;

  const int tid = threadIdx.x;
  const int wave = tid >> 5, lane = tid & 31;
  const int lane15 = lane & 15, half = lane >> 4;

  if (tid < 32) {
    int r = row0 + tid;
    int valid = r < cnt;
    int t = lists[e * NTOK + (valid ? r : 0)];
    Toks[tid] = t;
    Scl[tid] = valid ? combine[(size_t)t * NEXP + e] : 0.0f;
  }
  __syncthreads();

  // Stage 32 gathered bf16 token rows into LDS via async DMA (16B per op).
#pragma unroll
  for (int i = 0; i < 16; ++i) {
    int idx = tid + i * 256;       // 0..4095
    int r = idx >> 7, seg = idx & 127;
    async_b128(&Xs[r][seg * 8], xb + (size_t)Toks[r] * DDIM + seg * 8);
  }
  wait_async0();
  __syncthreads();

  const int n_sub = wave & 1;      // GEMM1: which 16 h-columns
  const int khalf = wave >> 1;     // GEMM1: 4-way K split (256 each)

  v8f C[2][8] = {};                // 32 x (wave's 128 out-cols) accumulators

  const __bf16* w1e = w1b + (size_t)e * HDIM * DDIM;
  const __bf16* w2e = w2b + (size_t)e * DDIM * HDIM;
  const int kbase = khalf * 256;

  for (int h0 = 0; h0 < HDIM; h0 += 32) {
    __syncthreads();  // protect Hs/Red from previous iteration's readers
    if (tid == 0 && h0 + 32 < HDIM) {
      __builtin_prefetch(w1e + (size_t)(h0 + 32) * DDIM, 0, 1);
      __builtin_prefetch(w2e + h0 + 32, 0, 1);
    }

    // -------- GEMM1 B prefetch: all 8 k-step fragments in distinct regs --------
    const __bf16* w1col = w1e + (size_t)(h0 + n_sub * 16 + lane15) * DDIM;
    v8bf B1lo[8], B1hi[8];
#pragma unroll
    for (int ks = 0; ks < 8; ++ks) {
      const __bf16* p = w1col + kbase + ks * 32 + half * 8;
      B1lo[ks] = *(const v8bf*)(p);
      B1hi[ks] = *(const v8bf*)(p + 16);
    }

    // -------- GEMM1: h[32x32] = Xs[32x1024] * W1[:, h0..h0+31], K-split --------
    v8f a1[2] = {};
#pragma unroll
    for (int ks = 0; ks < 8; ++ks) {
      int k0 = kbase + ks * 32;
#pragma unroll
      for (int m = 0; m < 2; ++m) {
        const __bf16* arow = &Xs[m * 16 + lane15][k0];
        v8bf alo = *(const v8bf*)(arow + half * 8);
        v8bf ahi = *(const v8bf*)(arow + 16 + half * 8);
        a1[m] = wmma_bf16(alo, ahi, B1lo[ks], B1hi[ks], a1[m]);
      }
    }

    // -------- GEMM2 B prefetch: issue early so latency hides under reduce/SiLU --------
    v8bf B2lo[8], B2hi[8];
#pragma unroll
    for (int nt = 0; nt < 8; ++nt) {
      const __bf16* p = w2e + (size_t)(wave * 128 + nt * 16 + lane15) * HDIM + h0 + half * 8;
      B2lo[nt] = *(const v8bf*)(p);
      B2hi[nt] = *(const v8bf*)(p + 16);
    }

    // -------- K-split reduction + SiLU -> Hs (bf16) --------
    if (wave >= 2) {
#pragma unroll
      for (int m = 0; m < 2; ++m)
#pragma unroll
        for (int v = 0; v < 8; ++v)
          Red[wave - 2][m * 256 + v * 32 + lane] = a1[m][v];
    }
    __syncthreads();
    if (wave < 2) {
#pragma unroll
      for (int m = 0; m < 2; ++m)
#pragma unroll
        for (int v = 0; v < 8; ++v) {
          float s = a1[m][v];
          s += Red[n_sub][m * 256 + v * 32 + lane];
          s += Red[n_sub + 2][m * 256 + v * 32 + lane];
          s += Red[n_sub + 4][m * 256 + v * 32 + lane];
          s = s / (1.0f + __expf(-s));  // SiLU
          Hs[m * 16 + half * 8 + v][n_sub * 16 + lane15] = (__bf16)s;
        }
    }
    __syncthreads();

    // -------- GEMM2: C += Hs[32x32] * W2[h0..h0+31, wave*128 .. +128] --------
    v8bf halo[2], hahi[2];
#pragma unroll
    for (int m = 0; m < 2; ++m) {
      const __bf16* hrow = &Hs[m * 16 + lane15][0];
      halo[m] = *(const v8bf*)(hrow + half * 8);
      hahi[m] = *(const v8bf*)(hrow + 16 + half * 8);
    }
#pragma unroll
    for (int nt = 0; nt < 8; ++nt)
#pragma unroll
      for (int m = 0; m < 2; ++m)
        C[m][nt] = wmma_bf16(halo[m], hahi[m], B2lo[nt], B2hi[nt], C[m][nt]);
  }

  // -------- epilogue: out[t, col] += gate * C (hw f32 atomics; 2 contributors) --------
  int   tReg[2][8];
  float sReg[2][8];
#pragma unroll
  for (int m = 0; m < 2; ++m)
#pragma unroll
    for (int v = 0; v < 8; ++v) {
      int mrow = m * 16 + half * 8 + v;
      tReg[m][v] = Toks[mrow];
      sReg[m][v] = Scl[mrow];
    }
#pragma unroll
  for (int m = 0; m < 2; ++m)
#pragma unroll
    for (int v = 0; v < 8; ++v) {
      float* orow = out + (size_t)tReg[m][v] * DDIM + wave * 128 + lane15;
      float s = sReg[m][v];
#pragma unroll
      for (int nt = 0; nt < 8; ++nt)
        unsafeAtomicAdd(orow + nt * 16, s * C[m][nt][v]);
    }
}

// ---------------- host launcher ----------------
extern "C" void kernel_launch(void* const* d_in, const int* in_sizes, int n_in,
                              void* d_out, int out_size, void* d_ws, size_t ws_size,
                              hipStream_t stream) {
  const float* x  = (const float*)d_in[0];   // [N, D]
  const float* gw = (const float*)d_in[1];   // [D, E]
  const float* w1 = (const float*)d_in[2];   // [E, D, H]
  const float* w2 = (const float*)d_in[3];   // [E, H, D]
  float* out = (float*)d_out;
  char* ws = (char*)d_ws;

  float*    combine = (float*)(ws + 0);               // 512 KB
  int*      counts  = (int*)(ws + 524288);            // 256 B slot
  int*      lists   = (int*)(ws + 524544);            // 512 KB
  uint16_t* xb      = (uint16_t*)(ws + 1048832);      // 32 MB bf16 x
  uint16_t* w1b     = (uint16_t*)(ws + 34603264);     // 64 MB bf16 w1^T
  uint16_t* w2b     = (uint16_t*)(ws + 101712128);    // 64 MB bf16 w2^T

  // zero output (atomic-accumulated) and per-expert counts
  k_zero_f4<<<16384, 256, 0, stream>>>((float4*)out);
  k_zero_counts<<<1, 64, 0, stream>>>(counts);

  // one-time precision/layout conversion
  k_cvt<<<16384, 256, 0, stream>>>(x, xb);            // 16,777,216 elems / 4
  dim3 tb(32, 8);
  k_tcvt<<<dim3(128, 32, NEXP), tb, 0, stream>>>(w1, w1b, DDIM, HDIM);  // [D,H]->[H,D]
  k_tcvt<<<dim3(32, 128, NEXP), tb, 0, stream>>>(w2, w2b, HDIM, DDIM);  // [H,D]->[D,H]

  // routing + scatter
  k_router<<<NTOK / 8, 256, 0, stream>>>(x, gw, combine, counts, lists);

  // fused expert FFN (worst-case tiles per expert; early-exit past counts[e])
  k_moe<<<NEXP * 512, 256, 0, stream>>>(xb, w1b, w2b, combine, counts, lists, out);
}
